// HorizonSpatialCrossAttention_82884278879241
// MI455X (gfx1250) — matrix operand order repaired and checked by reading, not compiled, not run on hardware
//
#include <hip/hip_runtime.h>
#include <hip/hip_bf16.h>

// Problem constants (match the reference)
#define NCAM   6
#define BH     100
#define BW     100
#define CCH    256
#define HGRID  100
#define WGRID  40
#define HF     48
#define WF     80
#define HEADS  8
#define PTS    8
#define HD     32
#define NQ     4000           // HGRID*WGRID
#define HWV    (HF * WF)      // 3840
#define KDIM   256
#define BPAD   72             // LDS row stride in halves (64 data + 8 pad)

typedef __attribute__((ext_vector_type(16))) _Float16 v16h;
typedef __attribute__((ext_vector_type(8)))  _Float16 v8h;
typedef __attribute__((ext_vector_type(8)))  float    v8f;

// ---------------------------------------------------------------------------
// f16-in / f32-out GEMM via v_wmma_f32_16x16x32_f16 (wave32).
// Block = 128 threads = 4 waves covering 4 consecutive M-tiles, all sharing
// one 64-column panel of B.  The full 256x64 B panel (32KB) is staged into
// LDS once per block (cooperative fill + single barrier), then the K-loop
// reads fragments with per-element 16-bit LDS loads (padded stride -> minimal bank conflict).
// Each wave computes a 16x64 tile (4 accumulators); K=256 fully unrolled
// -> 32 WMMAs per wave, zero cvt in the hot loop.
// grid.x = N/64, grid.y = ceil((M/16)/4)
// ---------------------------------------------------------------------------
__global__ void gemm_wmma_f16(const _Float16* __restrict__ A,
                              const _Float16* __restrict__ B,
                              const float* __restrict__ bias,
                              const float* __restrict__ resid,
                              float* __restrict__ C,
                              int M, int N) {
    __shared__ _Float16 Bs[KDIM * BPAD];         // 18,432 halves = 36KB

    const int tid    = threadIdx.x;
    const int lane   = tid & 31;
    const int wave   = tid >> 5;
    const int mtile  = blockIdx.y * 4 + wave;
    const int ngroup = blockIdx.x;               // 64 columns per group
    const int colg   = ngroup * 64;

    // ---- cooperative B-panel fill: 2048 chunks of 8 halves (16B) ----------
#pragma unroll
    for (int c = tid; c < KDIM * 8; c += 128) {
        const int k = c >> 3, cc = c & 7;
        *(v8h*)(&Bs[k * BPAD + cc * 8]) =
            *(const v8h*)(B + (size_t)k * N + colg + cc * 8);
    }
    __syncthreads();

    if (mtile * 16 >= M) return;                 // wave-uniform guard

    const int row0 = mtile * 16;
    const int m    = lane & 15;
    const int hi   = lane >> 4;

    const _Float16* __restrict__ arow = A + (size_t)(row0 + m) * KDIM;

    v8f acc0 = {}, acc1 = {}, acc2 = {}, acc3 = {};
#pragma unroll
    for (int k0 = 0; k0 < KDIM; k0 += 32) {
        // A fragment: K segments [8*hi, +8) and [16+8*hi, +8)  (two b128 loads)
        v8h alo = *(const v8h*)(arow + k0 + 8 * hi);
        v8h ahi = *(const v8h*)(arow + k0 + 16 + 8 * hi);
        v16h a = __builtin_shufflevector(alo, ahi,
                 0,1,2,3,4,5,6,7,8,9,10,11,12,13,14,15);
        // B fragments from LDS: lane holds column N=m, K segments [8*hi, +8) and [16+8*hi, +8)
        v16h b0, b1, b2, b3;
#pragma unroll
        for (int s = 0; s < 16; ++s) {
            const int kk = k0 + ((s < 8) ? (8 * hi + s) : (16 + 8 * hi + (s - 8)));
            const _Float16* br = &Bs[(size_t)kk * BPAD + m];
            b0[s] = br[0]; b1[s] = br[16]; b2[s] = br[32]; b3[s] = br[48];
        }

        acc0 = __builtin_amdgcn_wmma_f32_16x16x32_f16(false, a, false, b0, (short)0, acc0, false, false);
        acc1 = __builtin_amdgcn_wmma_f32_16x16x32_f16(false, a, false, b1, (short)0, acc1, false, false);
        acc2 = __builtin_amdgcn_wmma_f32_16x16x32_f16(false, a, false, b2, (short)0, acc2, false, false);
        acc3 = __builtin_amdgcn_wmma_f32_16x16x32_f16(false, a, false, b3, (short)0, acc3, false, false);
    }

    // D layout: lanes 0-15 -> N=lane, VGPR r -> M=r; lanes 16-31 -> M=8+r
#pragma unroll
    for (int t = 0; t < 4; ++t) {
        const v8f acc = (t == 0) ? acc0 : (t == 1) ? acc1 : (t == 2) ? acc2 : acc3;
        const int nn = colg + t * 16 + m;
        const float bv = bias ? bias[nn] : 0.0f;
#pragma unroll
        for (int r = 0; r < 8; ++r) {
            const int mm = row0 + r + 8 * hi;
            const size_t idx = (size_t)mm * N + nn;
            float v = acc[r] + bv;
            if (resid) v += resid[idx];
            C[idx] = v;
        }
    }
}

// generic f32 -> f16 conversion (weights)
__global__ void cvt_f16_kernel(const float* __restrict__ src,
                               _Float16* __restrict__ dst, int n) {
    const int i = blockIdx.x * blockDim.x + threadIdx.x;
    if (i < n) dst[i] = (_Float16)src[i];
}

// value gather + f16 convert: vsrc[cam*HWV+hw, ch] = value[hw, cam, ch]
__global__ void vgather_kernel(const float* __restrict__ value,
                               _Float16* __restrict__ vsrc) {
    const int b  = blockIdx.x;               // cam*HWV + hw
    const int ch = threadIdx.x;
    const int hw = b % HWV, cam = b / HWV;
    vsrc[(size_t)b * CCH + ch] =
        (_Float16)value[((size_t)hw * NCAM + cam) * CCH + ch];
}

// q_reb (f16) = nearest_sample(query+query_pos) per camera grid
__global__ void rebatch_kernel(const float* __restrict__ query,
                               const float* __restrict__ qpos,
                               const float* __restrict__ grid,
                               _Float16* __restrict__ q_reb) {
    const int b  = blockIdx.x;        // cam*NQ + (hg*WGRID + wg)
    const int ch = threadIdx.x;
    const float gx = grid[(size_t)b * 2 + 0];
    const float gy = grid[(size_t)b * 2 + 1];
    const float ix = rintf((gx + 1.0f) * 0.5f * (float)(BW - 1));
    const float iy = rintf((gy + 1.0f) * 0.5f * (float)(BH - 1));
    const bool valid = (ix >= 0.0f) && (ix <= (float)(BW - 1)) &&
                       (iy >= 0.0f) && (iy <= (float)(BH - 1));
    const int xc = min(max((int)ix, 0), BW - 1);
    const int yc = min(max((int)iy, 0), BH - 1);
    const size_t src = ((size_t)yc * BW + xc) * CCH + ch;
    q_reb[(size_t)b * CCH + ch] =
        (_Float16)(valid ? (query[src] + qpos[src]) : 0.0f);
}

// softmax over PTS=8 for each (row, head); in-place on [24000 x 64] buffer
__global__ void softmax8_kernel(float* __restrict__ att) {
    const int pair = blockIdx.x * blockDim.x + threadIdx.x;   // < 192000
    float* p = att + (size_t)pair * PTS;
    float v[PTS], mx = -3.0e38f;
#pragma unroll
    for (int i = 0; i < PTS; ++i) { v[i] = p[i]; mx = fmaxf(mx, v[i]); }
    float s = 0.0f;
#pragma unroll
    for (int i = 0; i < PTS; ++i) { v[i] = __expf(v[i] - mx); s += v[i]; }
    const float inv = 1.0f / s;
#pragma unroll
    for (int i = 0; i < PTS; ++i) p[i] = v[i] * inv;
}

// ---------------------------------------------------------------------------
// Deformable bilinear sampling + attention-weighted reduction.
// One wave per (cam, query, head); lanes = 32 head-dims (coalesced 128B reads).
// ---------------------------------------------------------------------------
__global__ void deform_kernel(const float* __restrict__ vproj,
                              const float* __restrict__ off,
                              const float* __restrict__ attw,
                              const float* __restrict__ ref,
                              float* __restrict__ outp) {
    const int g    = blockIdx.x * 8 + (threadIdx.x >> 5);  // < 192000
    const int d    = threadIdx.x & 31;
    const int head = g & (HEADS - 1);
    const int row  = g >> 3;                // cam*NQ + q
    const int cam  = row / NQ;

    const float rx = ref[(size_t)row * 2 + 0];
    const float ry = ref[(size_t)row * 2 + 1];
    const float* __restrict__ offp = off  + (size_t)row * (HEADS * PTS * 2) + head * (PTS * 2);
    const float* __restrict__ wp   = attw + (size_t)row * (HEADS * PTS)     + head * PTS;
    const float* __restrict__ vbase = vproj + (size_t)cam * HWV * CCH + head * HD + d;

    float acc = 0.0f;
#pragma unroll
    for (int p = 0; p < PTS; ++p) {
        const float x = (rx + offp[p * 2 + 0] * (1.0f / WF)) * (float)WF - 0.5f;
        const float y = (ry + offp[p * 2 + 1] * (1.0f / HF)) * (float)HF - 0.5f;
        const float w = wp[p];
        const float xf = floorf(x), yf = floorf(y);
        const float fx = x - xf, fy = y - yf;
        const int x0 = (int)xf, y0 = (int)yf;
#pragma unroll
        for (int cy = 0; cy < 2; ++cy) {
            const int yi = y0 + cy;
            if (yi < 0 || yi > HF - 1) continue;            // wave-uniform
            const float wy = cy ? fy : (1.0f - fy);
#pragma unroll
            for (int cx = 0; cx < 2; ++cx) {
                const int xi = x0 + cx;
                if (xi < 0 || xi > WF - 1) continue;        // wave-uniform
                const float wx = cx ? fx : (1.0f - fx);
                acc += w * wy * wx * vbase[(size_t)(yi * WF + xi) * CCH];
            }
        }
    }
    outp[(size_t)row * CCH + head * HD + d] = acc;
}

// ---------------------------------------------------------------------------
// Restore BEV: nearest-sample the [600 x 40] per-channel map at grid rows
// y and y+BH, sum overlap copies, scale by pillar counts, emit f16 (GEMM-A).
// ---------------------------------------------------------------------------
__global__ void restore_kernel(const float* __restrict__ attn,
                               const float* __restrict__ grid,
                               const float* __restrict__ counts,
                               _Float16* __restrict__ slots) {
    const int pos = blockIdx.x;           // y*BW + x, < 10000
    const int ch  = threadIdx.x;
    const int y = pos / BW, x = pos % BW;
    float acc = 0.0f;
#pragma unroll
    for (int half = 0; half < 2; ++half) {
        const int r = y + half * BH;
        const float gx = grid[((size_t)r * BW + x) * 2 + 0];
        const float gy = grid[((size_t)r * BW + x) * 2 + 1];
        const float ix = rintf((gx + 1.0f) * 0.5f * (float)(WGRID - 1));
        const float iy = rintf((gy + 1.0f) * 0.5f * (float)(NCAM * HGRID - 1));
        const bool valid = (ix >= 0.0f) && (ix <= (float)(WGRID - 1)) &&
                           (iy >= 0.0f) && (iy <= (float)(NCAM * HGRID - 1));
        const int xc = min(max((int)ix, 0), WGRID - 1);
        const int yc = min(max((int)iy, 0), NCAM * HGRID - 1);
        const int cam = yc / HGRID, hg = yc % HGRID;
        const size_t q = (size_t)cam * NQ + (size_t)hg * WGRID + xc;
        const float v = attn[q * CCH + ch];
        acc += valid ? v : 0.0f;
    }
    slots[(size_t)pos * CCH + ch] = (_Float16)(acc * counts[pos]);
}

// ---------------------------------------------------------------------------
extern "C" void kernel_launch(void* const* d_in, const int* in_sizes, int n_in,
                              void* d_out, int out_size, void* d_ws, size_t ws_size,
                              hipStream_t stream) {
    (void)in_sizes; (void)n_in; (void)out_size; (void)ws_size;
    const float* query  = (const float*)d_in[0];
    const float* value  = (const float*)d_in[1];   // [3840, 6, 256]
    const float* qgrid  = (const float*)d_in[2];
    const float* rgrid  = (const float*)d_in[3];
    const float* ref    = (const float*)d_in[4];
    const float* counts = (const float*)d_in[5];
    const float* qpos   = (const float*)d_in[6];
    const float* Wv     = (const float*)d_in[7];
    const float* bv     = (const float*)d_in[8];
    const float* Woff   = (const float*)d_in[9];
    const float* boff   = (const float*)d_in[10];
    const float* Watt   = (const float*)d_in[11];
    const float* batt   = (const float*)d_in[12];
    const float* Wout   = (const float*)d_in[13];
    const float* bout   = (const float*)d_in[14];
    float* out = (float*)d_out;

    // Workspace layout (bytes)
    char* ws = (char*)d_ws;
    _Float16* q_reb  = (_Float16*)(ws);                    // 24000*256*2 = 12,288,000
    _Float16* vsrc   = (_Float16*)(ws + 12288000);         // 23040*256*2 = 11,796,480
    float*    vproj  = (float*)   (ws + 24084480);         // 23040*256*4 = 23,592,960
    float*    offb   = (float*)   (ws + 47677440);         // 24000*128*4 = 12,288,000
    float*    attb   = (float*)   (ws + 59965440);         // 24000* 64*4 =  6,144,000
    float*    attn_o = (float*)   (ws + 66109440);         // 24000*256*4 = 24,576,000
    _Float16* Wv_h   = (_Float16*)(ws + 90685440);         // 65536*2
    _Float16* Woff_h = (_Float16*)(ws + 90816512);         // 32768*2
    _Float16* Watt_h = (_Float16*)(ws + 90882048);         // 16384*2
    _Float16* Wout_h = (_Float16*)(ws + 90914816);         // 65536*2 -> end 91,045,888
    _Float16* slots  = (_Float16*)offb;                    // offb dead after deform

    // 0) weights -> f16 (once)
    cvt_f16_kernel<<<256, 256, 0, stream>>>(Wv,   Wv_h,   CCH * CCH);
    cvt_f16_kernel<<<128, 256, 0, stream>>>(Woff, Woff_h, CCH * HEADS * PTS * 2);
    cvt_f16_kernel<<< 64, 256, 0, stream>>>(Watt, Watt_h, CCH * HEADS * PTS);
    cvt_f16_kernel<<<256, 256, 0, stream>>>(Wout, Wout_h, CCH * CCH);

    // 1) rebatch queries per camera (nearest sample of query+query_pos) -> f16
    rebatch_kernel<<<NCAM * NQ, 256, 0, stream>>>(query, qpos, qgrid, q_reb);

    // 2) gather value to [cam*HWV+hw, ch] layout, f16
    vgather_kernel<<<NCAM * HWV, 256, 0, stream>>>(value, vsrc);

    // 3) value projection GEMM
    gemm_wmma_f16<<<dim3(4, (NCAM * HWV / 16) / 4), 128, 0, stream>>>(
        vsrc, Wv_h, bv, nullptr, vproj, NCAM * HWV, CCH);

    // 4) offset + attention-logit GEMMs
    gemm_wmma_f16<<<dim3(2, (NCAM * NQ / 16) / 4), 128, 0, stream>>>(
        q_reb, Woff_h, boff, nullptr, offb, NCAM * NQ, HEADS * PTS * 2);
    gemm_wmma_f16<<<dim3(1, (NCAM * NQ / 16) / 4), 128, 0, stream>>>(
        q_reb, Watt_h, batt, nullptr, attb, NCAM * NQ, HEADS * PTS);

    // 5) softmax over the 8 sampling points per (row, head)
    softmax8_kernel<<<(NCAM * NQ * HEADS) / 256, 256, 0, stream>>>(attb);

    // 6) deformable bilinear sampling + weighted reduction
    deform_kernel<<<(NCAM * NQ * HEADS) / 8, 256, 0, stream>>>(
        vproj, offb, attb, ref, attn_o);

    // 7) restore BEV (overlap-sum + pillar-count scale) -> f16
    restore_kernel<<<BH * BW, 256, 0, stream>>>(attn_o, rgrid, counts, slots);

    // 8) output projection + bias + residual (625 M-tiles -> guarded grid)
    gemm_wmma_f16<<<dim3(4, (BH * BW / 16 + 3) / 4), 128, 0, stream>>>(
        slots, Wout_h, bout, query, out, BH * BW, CCH);
}
